// CausalSelfAttention_64665027609030
// MI455X (gfx1250) — compile-verified, matrix operand order checked by reading, not run on hardware
//
#include <hip/hip_runtime.h>

typedef __attribute__((ext_vector_type(16))) _Float16 v16h;
typedef __attribute__((ext_vector_type(8)))  _Float16 v8h;
typedef __attribute__((ext_vector_type(4)))  _Float16 v4h;
typedef __attribute__((ext_vector_type(8)))  float    v8f;

union Frag16 { v16h f; v8h h[2]; };

#define B_  2
#define T_  2048
#define C_  1280
#define H_  10
#define D_  128
#define MT_ (B_ * T_)          // 4096 rows for the big GEMMs
#define NEGBIG (-1e9f)

// ---------------------------------------------------------------------------
// fp32 -> f16 conversion, 4 elements per thread (b128 in, b64 out)
// ---------------------------------------------------------------------------
__global__ __launch_bounds__(256)
void cvt_f32_f16_x4(const float4* __restrict__ s, v4h* __restrict__ d, int n4) {
  int i = blockIdx.x * 256 + threadIdx.x;
  if (i < n4) {
    float4 f = s[i];
    v4h h;
    h.x = (_Float16)f.x; h.y = (_Float16)f.y;
    h.z = (_Float16)f.z; h.w = (_Float16)f.w;
    d[i] = h;
  }
}

// ---------------------------------------------------------------------------
// GEMM: Y[m][n] = sum_k A[m][k] * W[n][k]   (A:[M,K] f16, W:[N,K] f16, Y f32)
// One wave -> 32x64 tile: 2 A-frags x 4 B-frags -> 8 WMMA accumulators.
// 8-wave block covers a 64x256 macro tile (waves share A rows via L0/L2).
// A-frag lane layout (ISA 7.12.2): row = lane%16, elem i -> k = 16*(i>>3) + 8*(lane>>4) + (i&7)
// B symmetric with col = lane%16 (B[k][n] = W[n][k] row reads).
// ---------------------------------------------------------------------------
__global__ __launch_bounds__(256)
void gemm_f16_wmma(const _Float16* __restrict__ A,
                   const _Float16* __restrict__ W,
                   float* __restrict__ Y,
                   int M, int N, int K) {
  const int lane = threadIdx.x & 31;
  const int wid  = threadIdx.x >> 5;
  const int g    = lane >> 4;
  const int l16  = lane & 15;
  const int wm   = wid & 1;              // 2 x 4 wave grid inside the block
  const int wn   = wid >> 1;
  const int m0   = blockIdx.x * 64 + wm * 32;
  const int n0   = blockIdx.y * 256 + wn * 64;
  if (m0 >= M || n0 >= N) return;

  const _Float16* ar0 = A + (size_t)(m0 + l16) * K + 8 * g;
  const _Float16* ar1 = ar0 + (size_t)16 * K;
  const _Float16* wr0 = W + (size_t)(n0 + l16) * K + 8 * g;

  v8f acc[2][4] = {};
  for (int k = 0; k < K; k += 32) {
    Frag16 a[2], b[4];
    a[0].h[0] = *(const v8h*)(ar0 + k);
    a[0].h[1] = *(const v8h*)(ar0 + k + 16);
    a[1].h[0] = *(const v8h*)(ar1 + k);
    a[1].h[1] = *(const v8h*)(ar1 + k + 16);
#pragma unroll
    for (int j = 0; j < 4; ++j) {
      const _Float16* wr = wr0 + (size_t)(16 * j) * K + k;
      b[j].h[0] = *(const v8h*)(wr);
      b[j].h[1] = *(const v8h*)(wr + 16);
    }
#pragma unroll
    for (int i = 0; i < 2; ++i) {
#pragma unroll
      for (int j = 0; j < 4; ++j) {
        acc[i][j] = __builtin_amdgcn_wmma_f32_16x16x32_f16(
            false, a[i].f, false, b[j].f, (short)0, acc[i][j], false, false);
      }
    }
  }
  // C/D layout: lane holds col (n-tile + l16), rows (r + 8*g)
#pragma unroll
  for (int i = 0; i < 2; ++i) {
    float* yb = Y + (size_t)(m0 + 16 * i + 8 * g) * N + n0 + l16;
#pragma unroll
    for (int j = 0; j < 4; ++j) {
#pragma unroll
      for (int r = 0; r < 8; ++r)
        yb[(size_t)r * N + 16 * j] = acc[i][j][r];
    }
  }
}

// ---------------------------------------------------------------------------
// RoPE + RMSNorm on q,k; pack to [B,H,T,D] f16. v -> transposed [B,H,D,T] f16.
// One wave per (b,h,t) row of 128 elements.
// ---------------------------------------------------------------------------
__global__ __launch_bounds__(256)
void rope_rms_pack(const float* __restrict__ qb, const float* __restrict__ kb,
                   const float* __restrict__ vb, const float* __restrict__ cosp,
                   const float* __restrict__ sinp,
                   _Float16* __restrict__ Qh, _Float16* __restrict__ Kh,
                   _Float16* __restrict__ Vt) {
  const int lane = threadIdx.x & 31;
  const int wid  = threadIdx.x >> 5;
  const int row  = blockIdx.x * 8 + wid;     // (b*H + h)*T + t
  const int bh   = row / T_;
  const int t    = row % T_;
  const int b    = bh / H_;
  const int h    = bh % H_;
  const size_t src = ((size_t)b * T_ + t) * C_ + (size_t)h * D_;
  const float* cs = cosp + (size_t)t * (D_ / 2);
  const float* sn = sinp + (size_t)t * (D_ / 2);

#pragma unroll
  for (int which = 0; which < 2; ++which) {
    const float* x = (which == 0 ? qb : kb) + src;
    _Float16* outp = (which == 0 ? Qh : Kh) + ((size_t)bh * T_ + t) * D_;
    float y[4];
    float ss = 0.f;
#pragma unroll
    for (int j = 0; j < 2; ++j) {
      int p = lane + 32 * j;                 // pair index 0..63
      float x1 = x[p], x2 = x[p + 64];
      float c = cs[p], s = sn[p];
      float y1 = x1 * c + x2 * s;
      float y2 = -x1 * s + x2 * c;
      y[2 * j] = y1; y[2 * j + 1] = y2;
      ss += y1 * y1 + y2 * y2;
    }
    for (int off = 1; off < 32; off <<= 1) ss += __shfl_xor(ss, off, 32);
    float rn = rsqrtf(ss * (1.0f / D_) + 1e-5f);
#pragma unroll
    for (int j = 0; j < 2; ++j) {
      int p = lane + 32 * j;
      outp[p]      = (_Float16)(y[2 * j] * rn);
      outp[p + 64] = (_Float16)(y[2 * j + 1] * rn);
    }
  }
  // V: plain f16 convert + transpose for contiguous P@V B-fragments
  const float* xv = vb + src;
  _Float16* vt = Vt + (size_t)bh * D_ * T_;
#pragma unroll
  for (int j = 0; j < 4; ++j) {
    int d = lane + 32 * j;
    vt[(size_t)d * T_ + t] = (_Float16)xv[d];
  }
}

// ---------------------------------------------------------------------------
// Flash attention: one wave owns 16 query rows of one (b,h).
// Iterate 32-key tiles up to causal diagonal: QK^T wmma -> online softmax ->
// P re-layout via per-wave LDS -> P@V wmma into 16x128 f32 accumulator.
// ---------------------------------------------------------------------------
__global__ __launch_bounds__(256)
void attn_flash(const _Float16* __restrict__ Qh, const _Float16* __restrict__ Kh,
                const _Float16* __restrict__ Vt, _Float16* __restrict__ Yb) {
  __shared__ _Float16 plds[8][16 * 32];      // 1KB staging per wave
  const int lane = threadIdx.x & 31;
  const int wid  = threadIdx.x >> 5;
  const int g    = lane >> 4;
  const int l16  = lane & 15;
  const int bh   = blockIdx.y;
  const int b    = bh / H_;
  const int h    = bh % H_;
  const int q0   = (blockIdx.x * 8 + wid) * 16;

  const _Float16* Qb = Qh + ((size_t)bh * T_ + q0) * D_;
  const _Float16* Kb = Kh + (size_t)bh * T_ * D_;
  const _Float16* Vb = Vt + (size_t)bh * D_ * T_;

  // Q fragments: 4 chunks of 32 along D
  Frag16 qf[4];
#pragma unroll
  for (int c = 0; c < 4; ++c) {
    const _Float16* p = Qb + (size_t)l16 * D_ + 32 * c + 8 * g;
    qf[c].h[0] = *(const v8h*)(p);
    qf[c].h[1] = *(const v8h*)(p + 16);
  }

  v8f o[8];
#pragma unroll
  for (int i = 0; i < 8; ++i) o[i] = (v8f){};
  float mrow[8], lrow[8];
#pragma unroll
  for (int r = 0; r < 8; ++r) { mrow[r] = -1e30f; lrow[r] = 0.f; }

  const float scale = 0.088388347648318447f;  // 1/sqrt(128)
  _Float16* myP = &plds[wid][0];
  const int kend = q0 + 16;                   // exclusive key bound (causal)

  for (int k0 = 0; k0 < kend; k0 += 32) {
    if (k0 + 32 < kend)
      __builtin_prefetch(Kb + (size_t)(k0 + 32 + l16) * D_, 0, 1);

    // ---- scores: S0 (keys k0..k0+15), S1 (keys k0+16..k0+31)
    v8f s0 = {}, s1 = {};
#pragma unroll
    for (int c = 0; c < 4; ++c) {
      Frag16 kf;
      const _Float16* p = Kb + (size_t)(k0 + l16) * D_ + 32 * c + 8 * g;
      kf.h[0] = *(const v8h*)p;
      kf.h[1] = *(const v8h*)(p + 16);
      s0 = __builtin_amdgcn_wmma_f32_16x16x32_f16(false, qf[c].f, false, kf.f,
                                                  (short)0, s0, false, false);
    }
#pragma unroll
    for (int c = 0; c < 4; ++c) {
      Frag16 kf;
      const _Float16* p = Kb + (size_t)(k0 + 16 + l16) * D_ + 32 * c + 8 * g;
      kf.h[0] = *(const v8h*)p;
      kf.h[1] = *(const v8h*)(p + 16);
      s1 = __builtin_amdgcn_wmma_f32_16x16x32_f16(false, qf[c].f, false, kf.f,
                                                  (short)0, s1, false, false);
    }

    // ---- scale + causal mask + row max (rows confined to a 16-lane half)
    float smax[8];
#pragma unroll
    for (int r = 0; r < 8; ++r) {
      int rowq = q0 + r + 8 * g;
      float a = s0[r] * scale + ((k0 + l16)      > rowq ? NEGBIG : 0.f);
      float c = s1[r] * scale + ((k0 + 16 + l16) > rowq ? NEGBIG : 0.f);
      s0[r] = a; s1[r] = c;
      smax[r] = fmaxf(a, c);
    }
    for (int off = 1; off < 16; off <<= 1) {
#pragma unroll
      for (int r = 0; r < 8; ++r)
        smax[r] = fmaxf(smax[r], __shfl_xor(smax[r], off, 32));
    }

    // ---- online softmax update
    float rs[8];
#pragma unroll
    for (int r = 0; r < 8; ++r) {
      float mn   = fmaxf(mrow[r], smax[r]);
      float corr = __expf(mrow[r] - mn);
      mrow[r] = mn;
      float p0 = __expf(s0[r] - mn);
      float p1 = __expf(s1[r] - mn);
      s0[r] = p0; s1[r] = p1;
      rs[r] = p0 + p1;
      lrow[r] *= corr;
#pragma unroll
      for (int nc = 0; nc < 8; ++nc) o[nc][r] *= corr;
    }
    for (int off = 1; off < 16; off <<= 1) {
#pragma unroll
      for (int r = 0; r < 8; ++r) rs[r] += __shfl_xor(rs[r], off, 32);
    }
#pragma unroll
    for (int r = 0; r < 8; ++r) lrow[r] += rs[r];

    // ---- re-layout P (C-frag -> A-frag) through per-wave LDS tile
#pragma unroll
    for (int r = 0; r < 8; ++r) {
      myP[(r + 8 * g) * 32 + l16]      = (_Float16)s0[r];
      myP[(r + 8 * g) * 32 + 16 + l16] = (_Float16)s1[r];
    }
    asm volatile("" ::: "memory");  // keep ds store->load program order
    Frag16 pf;
    pf.h[0] = *(const v8h*)(myP + l16 * 32 + 8 * g);
    pf.h[1] = *(const v8h*)(myP + l16 * 32 + 16 + 8 * g);

    // ---- O += P @ V  (8 output chunks of 16 along D)
#pragma unroll
    for (int nc = 0; nc < 8; ++nc) {
      Frag16 vf;
      const _Float16* p = Vb + (size_t)(nc * 16 + l16) * T_ + k0 + 8 * g;
      vf.h[0] = *(const v8h*)p;
      vf.h[1] = *(const v8h*)(p + 16);
      o[nc] = __builtin_amdgcn_wmma_f32_16x16x32_f16(false, pf.f, false, vf.f,
                                                     (short)0, o[nc], false, false);
    }
  }

  // ---- normalize and store y as f16 [B,T,C] for the projection GEMM
  _Float16* yb = Yb + ((size_t)b * T_ + q0 + 8 * g) * C_ + (size_t)h * D_ + l16;
#pragma unroll
  for (int r = 0; r < 8; ++r) {
    float inv = 1.0f / lrow[r];
#pragma unroll
    for (int nc = 0; nc < 8; ++nc)
      yb[(size_t)r * C_ + nc * 16] = (_Float16)(o[nc][r] * inv);
  }
}

// ---------------------------------------------------------------------------
extern "C" void kernel_launch(void* const* d_in, const int* in_sizes, int n_in,
                              void* d_out, int out_size, void* d_ws, size_t ws_size,
                              hipStream_t stream) {
  (void)in_sizes; (void)n_in; (void)out_size; (void)ws_size;
  const float* x    = (const float*)d_in[0];
  const float* cosp = (const float*)d_in[1];
  const float* sinp = (const float*)d_in[2];
  const float* Wq   = (const float*)d_in[3];
  const float* Wk   = (const float*)d_in[4];
  const float* Wv   = (const float*)d_in[5];
  const float* Wp   = (const float*)d_in[6];
  float* out = (float*)d_out;

  char* ws = (char*)d_ws;
  size_t off = 0;
  auto alloc = [&](size_t bytes) {
    void* p = ws + off;
    off = (off + bytes + 255) & ~(size_t)255;
    return p;
  };

  const size_t actsH = (size_t)MT_ * C_;           // 4096*1280
  const size_t wH    = (size_t)C_ * C_;

  _Float16* xf16 = (_Float16*)alloc(actsH * 2);
  _Float16* wq16 = (_Float16*)alloc(wH * 2);
  _Float16* wk16 = (_Float16*)alloc(wH * 2);
  _Float16* wv16 = (_Float16*)alloc(wH * 2);
  _Float16* wp16 = (_Float16*)alloc(wH * 2);
  float*    qbuf = (float*)alloc(actsH * 4);
  float*    kbuf = (float*)alloc(actsH * 4);
  float*    vbuf = (float*)alloc(actsH * 4);
  _Float16* Qh   = (_Float16*)alloc(actsH * 2);
  _Float16* Kh   = (_Float16*)alloc(actsH * 2);
  _Float16* Vt   = (_Float16*)alloc(actsH * 2);
  _Float16* Yb   = (_Float16*)alloc(actsH * 2);

  // 1) f32 -> f16 conversions (vectorized x4)
  cvt_f32_f16_x4<<<(int)((actsH / 4 + 255) / 256), 256, 0, stream>>>(
      (const float4*)x, (v4h*)xf16, (int)(actsH / 4));
  cvt_f32_f16_x4<<<(int)((wH / 4 + 255) / 256), 256, 0, stream>>>(
      (const float4*)Wq, (v4h*)wq16, (int)(wH / 4));
  cvt_f32_f16_x4<<<(int)((wH / 4 + 255) / 256), 256, 0, stream>>>(
      (const float4*)Wk, (v4h*)wk16, (int)(wH / 4));
  cvt_f32_f16_x4<<<(int)((wH / 4 + 255) / 256), 256, 0, stream>>>(
      (const float4*)Wv, (v4h*)wv16, (int)(wH / 4));
  cvt_f32_f16_x4<<<(int)((wH / 4 + 255) / 256), 256, 0, stream>>>(
      (const float4*)Wp, (v4h*)wp16, (int)(wH / 4));

  // 2) QKV projections (y = x @ W^T), 32x64 per wave, 64x256 per block
  dim3 ggrid(MT_ / 64, C_ / 256);                  // 64 x 5
  gemm_f16_wmma<<<ggrid, 256, 0, stream>>>(xf16, wq16, qbuf, MT_, C_, C_);
  gemm_f16_wmma<<<ggrid, 256, 0, stream>>>(xf16, wk16, kbuf, MT_, C_, C_);
  gemm_f16_wmma<<<ggrid, 256, 0, stream>>>(xf16, wv16, vbuf, MT_, C_, C_);

  // 3) RoPE + RMSNorm + repack / V transpose
  rope_rms_pack<<<(B_ * H_ * T_) / 8, 256, 0, stream>>>(qbuf, kbuf, vbuf, cosp,
                                                        sinp, Qh, Kh, Vt);

  // 4) flash attention
  dim3 agrid(T_ / (16 * 8), B_ * H_);              // 16 x 20
  attn_flash<<<agrid, 256, 0, stream>>>(Qh, Kh, Vt, Yb);

  // 5) output projection -> fp32 d_out
  gemm_f16_wmma<<<ggrid, 256, 0, stream>>>(Yb, wp16, out, MT_, C_, C_);
}
